// SSA_48816598286500
// MI455X (gfx1250) — matrix operand order repaired and checked by reading, not compiled, not run on hardware
//
#include <hip/hip_runtime.h>
#include <hip/hip_bf16.h>

typedef __bf16 bf16_t;
typedef __attribute__((ext_vector_type(16))) __bf16 bf16x16;
typedef __attribute__((ext_vector_type(4)))  __bf16 bf16x4;
typedef __attribute__((ext_vector_type(8)))  float  floatx8;
typedef __attribute__((ext_vector_type(4)))  float  floatx4;
typedef __attribute__((ext_vector_type(4)))  unsigned int u32x4;
typedef __attribute__((ext_vector_type(4)))  int i32x4;
typedef __attribute__((ext_vector_type(8)))  int i32x8;

#define B_  32
#define C_  256
#define N_  3136   // 56*56

// ---- Tensor Data Mover availability probes -------------------------------
#if defined(__has_builtin)
# if __has_builtin(__builtin_amdgcn_tensor_load_to_lds) && __has_builtin(__builtin_amdgcn_s_wait_tensorcnt)
#  define USE_TDM 1
# endif
#endif
#ifndef USE_TDM
# define USE_TDM 0
#endif
#if USE_TDM
# if __has_include(<hip/amd_detail/amd_gfx1250_TDM.h>)
#  define TDM_SIX_ARGS 1   // amdgpu-toolchain / therock headers -> 6-arg builtin
# else
#  define TDM_SIX_ARGS 0   // ROCm 7.2 -> 5-arg builtin
# endif
#endif

#if USE_TDM
// Issue one 2D TDM copy: tile_rows x tile_w (2-byte elems), global row stride
// row_stride_elems, into contiguous LDS at byte offset lds_off. (ISA 8.3-8.5)
__device__ __forceinline__ void tdm_load_2d(const void* gsrc, unsigned lds_off,
                                            unsigned tile_w, unsigned tile_rows,
                                            unsigned long long row_stride_elems) {
    unsigned long long ga = (unsigned long long)gsrc;
    u32x4 g0 = {};
    g0[0] = 1u;                                              // count=1, user D#
    g0[1] = lds_off;                                         // lds byte address
    g0[2] = (unsigned)ga;                                    // global addr [31:0]
    g0[3] = (unsigned)((ga >> 32) & 0x1FFFFFFull) | (2u << 30); // addr[56:32] | type=2

    // group1 bit layout: [17:16] data_size(1->2B) | [79:48] tensor_dim0 |
    // [111:80] tensor_dim1 | [127:112] tile_dim0 | [143:128] tile_dim1 |
    // [207:160] tensor_dim0_stride
    unsigned long long q0, q1, q2, q3;
    q0  = (1ull << 16);
    q0 |= ((unsigned long long)tile_w & 0xFFFFull) << 48;
    q1  = ((unsigned long long)tile_w >> 16) & 0xFFFFull;
    q1 |= ((unsigned long long)tile_rows) << 16;
    q1 |= ((unsigned long long)tile_w & 0xFFFFull) << 48;
    q2  = ((unsigned long long)tile_rows & 0xFFFFull);
    q2 |= (row_stride_elems & 0xFFFFFFFFull) << 32;
    q3  = (row_stride_elems >> 32) & 0xFFFFull;

    i32x8 g1;
    g1[0] = (int)(unsigned)q0; g1[1] = (int)(q0 >> 32);
    g1[2] = (int)(unsigned)q1; g1[3] = (int)(q1 >> 32);
    g1[4] = (int)(unsigned)q2; g1[5] = (int)(q2 >> 32);
    g1[6] = (int)(unsigned)q3; g1[7] = (int)(q3 >> 32);
    i32x4 z4 = {};
#if TDM_SIX_ARGS
    i32x8 z8 = {};
    __builtin_amdgcn_tensor_load_to_lds(g0, g1, z4, z4, z8, 0);
#else
    __builtin_amdgcn_tensor_load_to_lds(g0, g1, z4, z4, 0);
#endif
}
#endif

// ---------------------------------------------------------------------------
// A (16x32 MxK) bf16 fragment: lane<16 -> row M=lane,   K=k0+0..15
//                              lane>=16 -> row M=lane-16, K=k0+16..31
// ---------------------------------------------------------------------------
__device__ __forceinline__ bf16x16 load_a(const bf16_t* __restrict__ A, int lda,
                                          int mbase, int k0, int lane) {
    int row  = mbase + (lane & 15);
    int koff = k0 + ((lane >> 4) << 4);
    const bf16_t* p = A + (long)row * lda + koff;
    bf16x16 a;
#pragma unroll
    for (int i = 0; i < 16; ++i) a[i] = p[i];
    return a;
}

// B (32x16 KxN) fragment from q rows (B[k][col] == Q[col][k]): contiguous.
__device__ __forceinline__ bf16x16 load_b_rows(const bf16_t* __restrict__ Q, int ldq,
                                               int nbase, int k0, int lane) {
    int col  = nbase + (lane & 15);
    int koff = k0 + ((lane >> 4) << 4);
    const bf16_t* p = Q + (long)col * ldq + koff;
    bf16x16 b;
#pragma unroll
    for (int i = 0; i < 16; ++i) b[i] = p[i];
    return b;
}

// ---------------------------------------------------------------------------
// fp32 -> bf16 cast, 4 elements/thread
// ---------------------------------------------------------------------------
__global__ void cast_f32_bf16(const float* __restrict__ in, bf16_t* __restrict__ o) {
    long i = ((long)blockIdx.x * blockDim.x + threadIdx.x) * 4;
    floatx4 v = *(const floatx4*)(in + i);
    bf16x4 r = { (bf16_t)v.x, (bf16_t)v.y, (bf16_t)v.z, (bf16_t)v.w };
    *(bf16x4*)(o + i) = r;
}

// ---------------------------------------------------------------------------
// Strided-B GEMM (conv1 / apply / conv2), all-bf16 inputs, K = 256.
//   Out[b][m][n] = sum_k A[b][m][k] * Bsrc[b][k][n]  (+ bias[m])
// Block = 128 threads = 4 waves. Block tile 64x64; wave tile 16x64 (4 accs).
// B tile [32k x 64n] staged row-major into LDS by TDM (or manual fallback).
// ---------------------------------------------------------------------------
template <typename TOUT, bool ADD_BIAS>
__global__ void gemm_tiled(const bf16_t* __restrict__ A, long aBatchStride,
                           const bf16_t* __restrict__ Bsrc,
                           TOUT* __restrict__ Out,
                           const float* __restrict__ bias) {
    __shared__ bf16_t Bt[32][64];                 // row-major [k][n], 4 KB
    const int lane = threadIdx.x & 31;
    const int wave = threadIdx.x >> 5;
    const int b    = blockIdx.z;
    const int n0g  = blockIdx.x * 64;
    const int m0   = (blockIdx.y * 4 + wave) * 16;

    const bf16_t* Ab = A    + (long)b * aBatchStride;
    const bf16_t* Bb = Bsrc + (long)b * C_ * N_;
    TOUT*         Ob = Out  + (long)b * C_ * N_;

    floatx8 acc[4] = {};
    for (int k0 = 0; k0 < C_; k0 += 32) {
        __syncthreads();
#if USE_TDM
        if (threadIdx.x == 0) {
            tdm_load_2d(Bb + (long)k0 * N_ + n0g,
                        (unsigned)(unsigned long long)(const void*)&Bt[0][0],
                        /*tile_w=*/64, /*tile_rows=*/32, /*stride=*/N_);
            __builtin_amdgcn_s_wait_tensorcnt(0);
        }
#else
        for (int idx = threadIdx.x; idx < 2048; idx += 128) {
            int kk = idx >> 6, nn = idx & 63;
            Bt[kk][nn] = Bb[(long)(k0 + kk) * N_ + n0g + nn];
        }
#endif
        __syncthreads();

        bf16x16 a = load_a(Ab, C_, m0, k0, lane);
        const int kh  = (lane >> 4) << 4;
        const int c16 = lane & 15;
#pragma unroll
        for (int s = 0; s < 4; ++s) {
            bf16x16 bfrag;
            const int col = s * 16 + c16;
#pragma unroll
            for (int i = 0; i < 16; ++i) bfrag[i] = Bt[kh + i][col];
            acc[s] = __builtin_amdgcn_wmma_f32_16x16x32_bf16(false, a, false, bfrag,
                                                             (short)0, acc[s], false, false);
        }
    }

    const int c16   = lane & 15;
    const int rbase = m0 + ((lane >> 4) << 3);
#pragma unroll
    for (int s = 0; s < 4; ++s) {
        const int ncol = n0g + s * 16 + c16;
#pragma unroll
        for (int r = 0; r < 8; ++r) {
            float v = acc[s][r];
            if (ADD_BIAS) v += bias[rbase + r];
            Ob[(long)(rbase + r) * N_ + ncol] = (TOUT)v;
        }
    }
}

// ---------------------------------------------------------------------------
// scores[b] = q[b] x q[b]^T : both fragments contiguous. M=N=256, K=3136.
// Wave tile 16x64 (4 accs), one A fragment per K-step shared by 4 WMMAs.
// ---------------------------------------------------------------------------
__global__ void gemm_qqt(const bf16_t* __restrict__ Q, float* __restrict__ S) {
    const int lane = threadIdx.x & 31;
    const int wave = threadIdx.x >> 5;
    const int b    = blockIdx.z;
    const int n0   = blockIdx.x * 64;
    const int m0   = (blockIdx.y * 4 + wave) * 16;

    const bf16_t* Qb = Q + (long)b * C_ * N_;
    float*        Sb = S + (long)b * C_ * C_;

    floatx8 acc[4] = {};
    for (int k0 = 0; k0 < N_; k0 += 32) {
        bf16x16 a = load_a(Qb, N_, m0, k0, lane);
#pragma unroll
        for (int s = 0; s < 4; ++s) {
            bf16x16 bb = load_b_rows(Qb, N_, n0 + s * 16, k0, lane);
            acc[s] = __builtin_amdgcn_wmma_f32_16x16x32_bf16(false, a, false, bb,
                                                             (short)0, acc[s], false, false);
        }
    }
    const int c16   = lane & 15;
    const int rbase = m0 + ((lane >> 4) << 3);
#pragma unroll
    for (int s = 0; s < 4; ++s) {
        const int ncol = n0 + s * 16 + c16;
#pragma unroll
        for (int r = 0; r < 8; ++r)
            Sb[(long)(rbase + r) * C_ + ncol] = acc[s][r];
    }
}

// ---------------------------------------------------------------------------
// Row softmax over 256 channels, fp32 -> bf16. One wave per row (wave32).
// ---------------------------------------------------------------------------
__global__ void softmax_rows(const float* __restrict__ S, bf16_t* __restrict__ A) {
    const int lane = threadIdx.x & 31;
    const int wave = threadIdx.x >> 5;
    const long row = (long)blockIdx.x * 8 + wave;
    const float* p = S + row * C_;

    float v[8];
    float mx = -3.4e38f;
#pragma unroll
    for (int i = 0; i < 8; ++i) { v[i] = p[i * 32 + lane]; mx = fmaxf(mx, v[i]); }
#pragma unroll
    for (int off = 16; off > 0; off >>= 1) mx = fmaxf(mx, __shfl_xor(mx, off, 32));
    float sum = 0.f;
#pragma unroll
    for (int i = 0; i < 8; ++i) { v[i] = __expf(v[i] - mx); sum += v[i]; }
#pragma unroll
    for (int off = 16; off > 0; off >>= 1) sum += __shfl_xor(sum, off, 32);
    const float inv = __frcp_rn(sum);

    bf16_t* o = A + row * C_;
#pragma unroll
    for (int i = 0; i < 8; ++i) o[i * 32 + lane] = (bf16_t)(v[i] * inv);
}

// ---------------------------------------------------------------------------
// Workspace (bytes), region R reused (xb -> scores -> o2, lifetimes disjoint):
//   Wb   @ 0         : 131072
//   q    @ 131072    : 51380224
//   attn @ 51511296  : 4194304
//   R    @ 55705600  : 51380224   (xb bf16 | scores fp32 (8.4MB) | o2 bf16)
//   total 107085824  (~107 MB)
// ---------------------------------------------------------------------------
extern "C" void kernel_launch(void* const* d_in, const int* in_sizes, int n_in,
                              void* d_out, int out_size, void* d_ws, size_t ws_size,
                              hipStream_t stream) {
    const float* x    = (const float*)d_in[0];   // [32,256,56,56]
    const float* W    = (const float*)d_in[1];   // [256,256]
    const float* bias = (const float*)d_in[2];   // [256]
    float*       out  = (float*)d_out;

    char* ws = (char*)d_ws;
    bf16_t* Wb   = (bf16_t*)(ws);
    bf16_t* q    = (bf16_t*)(ws + 131072);
    bf16_t* attn = (bf16_t*)(ws + 51511296);
    bf16_t* xb   = (bf16_t*)(ws + 55705600);     // region R as x_bf16
    float*  sc   = (float*) (ws + 55705600);     // region R as scores
    bf16_t* o2   = (bf16_t*)(ws + 55705600);     // region R as attn@q

    // casts: W (65536 el) and x (25690112 el), 4 el/thread
    cast_f32_bf16<<<dim3(64),    256, 0, stream>>>(W, Wb);
    cast_f32_bf16<<<dim3(25088), 256, 0, stream>>>(x, xb);

    // conv1: q = W x + b
    gemm_tiled<bf16_t, true><<<dim3(49, 4, 32), 128, 0, stream>>>(
        Wb, 0L, xb, q, bias);

    // scores = q q^T   (xb dead; R becomes scores)
    gemm_qqt<<<dim3(4, 4, 32), 128, 0, stream>>>(q, sc);

    // attn = softmax(scores)
    softmax_rows<<<dim3(1024), 256, 0, stream>>>(sc, attn);

    // o2 = attn q      (scores dead; R becomes o2)
    gemm_tiled<bf16_t, false><<<dim3(49, 4, 32), 128, 0, stream>>>(
        attn, (long)C_ * C_, q, o2, bias);

    // out = W o2 + b
    gemm_tiled<float, true><<<dim3(49, 4, 32), 128, 0, stream>>>(
        Wb, 0L, o2, out, bias);
}
